// Encoder_GCN_67912022884657
// MI455X (gfx1250) — matrix operand order repaired and checked by reading, not compiled, not run on hardware
//
#include <hip/hip_runtime.h>
#include <hip/hip_bf16.h>

// ---------------------------------------------------------------------------
// CDNA5 (gfx1250) implementation: bf16 WMMA GEMMs (fp32 accumulate) + atomic
// scatter aggregation for the GCN message passing and the pair scatter-mean.
// ---------------------------------------------------------------------------

typedef __attribute__((ext_vector_type(16))) __bf16       v16bf;
typedef __attribute__((ext_vector_type(8)))  float        v8f;
typedef __attribute__((ext_vector_type(4)))  unsigned int v4u;

union V16BF { v16bf bf; v4u q[2]; };

__device__ __forceinline__ unsigned short f2bf(float f) {
  union { float f; unsigned int u; } c; c.f = f;
  unsigned int u = c.u;
  u += 0x7FFFu + ((u >> 16) & 1u);   // round-to-nearest-even
  return (unsigned short)(u >> 16);
}

// ---------------------------------------------------------------------------
// Pack fp32 A [M x K] row-major into bf16 WMMA A-fragment layout:
//   Ap[((mt*ksteps + ks)*32 + lane)*16 + i]
//   lane<16: row=mt*16+lane,     K = ks*32 + {0..7, 16..23}[i]
//   lane>=16: row=mt*16+lane-16, K = ks*32 + {8..15,24..31}[i]
// Every GEMM A load becomes two contiguous 16B reads per lane.
// ---------------------------------------------------------------------------
__global__ void pack_a_kernel(const float* __restrict__ X,
                              unsigned short* __restrict__ Ap, int M, int K) {
  const int ksteps = K >> 5;
  const long total = (long)((M + 15) >> 4) * ksteps * 512;
  for (long idx = (long)blockIdx.x * blockDim.x + threadIdx.x; idx < total;
       idx += (long)gridDim.x * blockDim.x) {
    int i    = (int)(idx & 15);
    int lane = (int)((idx >> 4) & 31);
    long t   = idx >> 9;
    int ks   = (int)(t % ksteps);
    int mt   = (int)(t / ksteps);
    int row  = (mt << 4) + (lane & 15);
    if (row >= M) row = M - 1;                 // tail tile: duplicate last row
    int kbase = (lane >> 4) << 3;              // 0 or 8
    int k = (ks << 5) + kbase + (i < 8 ? i : i + 8);
    Ap[idx] = f2bf(X[(size_t)row * K + k]);
  }
}

// ---------------------------------------------------------------------------
// Pack fp32 W [K x N] row-major into bf16 WMMA B-fragment layout:
//   Bp[((nt*ksteps + ks)*32 + lane)*16 + i]
//   lane<16: col=nt*16+lane,     K = ks*32 + {0..7, 16..23}[i]
//   lane>=16: col=nt*16+lane-16, K = ks*32 + {8..15,24..31}[i]
// ---------------------------------------------------------------------------
__global__ void pack_w_kernel(const float* __restrict__ W,
                              unsigned short* __restrict__ Bp, int K, int N) {
  const int ksteps = K >> 5;
  const long total = (long)(N >> 4) * ksteps * 512;
  for (long idx = (long)blockIdx.x * blockDim.x + threadIdx.x; idx < total;
       idx += (long)gridDim.x * blockDim.x) {
    int i    = (int)(idx & 15);
    int lane = (int)((idx >> 4) & 31);
    long t   = idx >> 9;
    int ks   = (int)(t % ksteps);
    int nt   = (int)(t / ksteps);
    int col  = (nt << 4) + (lane & 15);
    int koff = (lane >> 4) << 3;               // 0 or 8
    int k = (ks << 5) + koff + (i < 8 ? i : i + 8);
    Bp[idx] = f2bf(W[(size_t)k * N + col]);
  }
}

// ---------------------------------------------------------------------------
// WMMA GEMM: C[M x N] = A[M x K] * W[K x N] (+ bias), bf16 in, f32 out.
// One wave computes a 16x64 strip: 4 accumulators share one A fragment,
// so per K-step: 5 x 32B loads -> 4 x v_wmma_f32_16x16x32_bf16.
// KSTEPS_CT != 0 makes the K loop fully unrolled with all load offsets
// folded into 24-bit instruction immediates off two scalar base pointers
// (global_load_b128, no per-iteration 64-bit VALU address math, no flat).
// Requires N % 64 == 0 (N = 256 here).
// ---------------------------------------------------------------------------
#define NTW 4

template <int KSTEPS_CT>
__global__ void __launch_bounds__(256)
wmma_gemm_kernel(const unsigned short* __restrict__ Ap,
                 const unsigned short* __restrict__ Bp,
                 const float* __restrict__ bias,
                 float* __restrict__ C, int M, int N, int ksteps_rt) {
  const int ksteps  = KSTEPS_CT ? KSTEPS_CT : ksteps_rt;
  const int lane    = threadIdx.x & 31;
  const int wib     = threadIdx.x >> 5;
  const int mtiles  = (M + 15) >> 4;
  const int ngroups = N >> 6;
  const int wid = blockIdx.x * (blockDim.x >> 5) + wib;
  if (wid >= mtiles * ngroups) return;
  const int mt = wid / ngroups;
  const int ng = wid % ngroups;

  // Single scalar bases; all per-step/per-tile displacements are constants
  // when KSTEPS_CT != 0, so loads use immediate offsets.
  const unsigned short* aP =
      Ap + ((size_t)mt * ksteps) * 512 + (size_t)lane * 16;
  const unsigned short* bP =
      Bp + ((size_t)ng * NTW * ksteps) * 512 + (size_t)lane * 16;

  v8f zero = {0.f, 0.f, 0.f, 0.f, 0.f, 0.f, 0.f, 0.f};
  v8f acc[NTW];
#pragma unroll
  for (int j = 0; j < NTW; ++j) acc[j] = zero;

#pragma unroll
  for (int ks = 0; ks < ksteps; ++ks) {
    V16BF a;
    a.q[0] = *(const v4u*)(aP + (size_t)ks * 512);
    a.q[1] = *(const v4u*)(aP + (size_t)ks * 512 + 8);
#pragma unroll
    for (int j = 0; j < NTW; ++j) {
      const unsigned short* p = bP + ((size_t)j * ksteps + ks) * 512;
      V16BF b;
      b.q[0] = *(const v4u*)(p);
      b.q[1] = *(const v4u*)(p + 8);
      acc[j] = __builtin_amdgcn_wmma_f32_16x16x32_bf16(
          false, a.bf, false, b.bf, (short)0, acc[j], false, false);
    }
  }

  // C/D layout: VGPR r, lanes 0-15 -> M=r, lanes 16-31 -> M=8+r; N = lane&15.
  const int rbase = (mt << 4) + ((lane >> 4) << 3);
  const int cl = lane & 15;
#pragma unroll
  for (int j = 0; j < NTW; ++j) {
    int col = ((ng * NTW + j) << 4) + cl;
    float bv = bias ? bias[col] : 0.0f;
#pragma unroll
    for (int r = 0; r < 8; ++r) {
      int row = rbase + r;
      if (row < M) C[(size_t)row * N + col] = acc[j][r] + bv;
    }
  }
}

// ------------------------- elementwise / scatter ---------------------------

__global__ void fill_kernel(float* __restrict__ p, float v, long n) {
  for (long i = (long)blockIdx.x * blockDim.x + threadIdx.x; i < n;
       i += (long)gridDim.x * blockDim.x) p[i] = v;
}

__global__ void deg_accum_kernel(const int* __restrict__ dst,
                                 float* __restrict__ deg, int E) {
  for (int i = blockIdx.x * blockDim.x + threadIdx.x; i < E;
       i += gridDim.x * blockDim.x) atomicAdd(&deg[dst[i]], 1.0f);
}

__global__ void rsqrt_kernel(float* __restrict__ p, int n) {
  for (int i = blockIdx.x * blockDim.x + threadIdx.x; i < n;
       i += gridDim.x * blockDim.x) p[i] = rsqrtf(p[i]);
}

// agg[r][:] = xw[r][:] * dinv[r]^2   (self-loop term; also initializes agg)
__global__ void self_term_kernel(const float* __restrict__ xw,
                                 const float* __restrict__ dinv,
                                 float* __restrict__ agg, int D) {
  int r = blockIdx.x, j = threadIdx.x;
  float di = dinv[r];
  size_t i = (size_t)r * D + j;
  agg[i] = xw[i] * di * di;
}

// one block per edge, 64 threads x float4: agg[d] += xw[s] * dinv[s]*dinv[d]
__global__ void edge_agg_kernel(const int* __restrict__ src,
                                const int* __restrict__ dst,
                                const float* __restrict__ xw,
                                const float* __restrict__ dinv,
                                float* __restrict__ agg, int D) {
  int e = blockIdx.x;
  int s = src[e], d = dst[e];
  float coef = dinv[s] * dinv[d];
  int c = threadIdx.x << 2;
  const float4 v = *(const float4*)(xw + (size_t)s * D + c);
  float* o = agg + (size_t)d * D + c;
  atomicAdd(o + 0, v.x * coef);
  atomicAdd(o + 1, v.y * coef);
  atomicAdd(o + 2, v.z * coef);
  atomicAdd(o + 3, v.w * coef);
}

__global__ void bias_elu_kernel(const float* __restrict__ agg,
                                const float* __restrict__ b,
                                float* __restrict__ out, int D) {
  int r = blockIdx.x, j = threadIdx.x;
  size_t i = (size_t)r * D + j;
  float x = agg[i] + b[j];
  out[i] = x > 0.f ? x : (expf(x) - 1.f);
}

__global__ void count_pairs_kernel(const int* __restrict__ v,
                                   float* __restrict__ counts, int P) {
  for (int i = blockIdx.x * blockDim.x + threadIdx.x; i < P;
       i += gridDim.x * blockDim.x) atomicAdd(&counts[v[i]], 1.0f);
}

// one block per pair, 64 threads x float4: xt[v] += h[u]
__global__ void pair_scatter_kernel(const int* __restrict__ u,
                                    const int* __restrict__ v,
                                    const float* __restrict__ h,
                                    float* __restrict__ xt, int D) {
  int p = blockIdx.x;
  int s = u[p], d = v[p];
  int c = threadIdx.x << 2;
  const float4 val = *(const float4*)(h + (size_t)s * D + c);
  float* o = xt + (size_t)d * D + c;
  atomicAdd(o + 0, val.x);
  atomicAdd(o + 1, val.y);
  atomicAdd(o + 2, val.z);
  atomicAdd(o + 3, val.w);
}

__global__ void finalize_xt_kernel(float* __restrict__ xt,
                                   const float* __restrict__ counts, int D) {
  int r = blockIdx.x, j = threadIdx.x;
  size_t i = (size_t)r * D + j;
  xt[i] = xt[i] / (1.0f + counts[r]);
}

// ---------------------------------------------------------------------------

static inline int nblk(long n, int t) {
  long b = (n + t - 1) / t;
  if (b > 262144) b = 262144;
  if (b < 1) b = 1;
  return (int)b;
}

static inline void launch_gemm(const unsigned short* Ap, const unsigned short* Bp,
                               const float* bias, float* C, int M, int N, int K,
                               hipStream_t stream) {
  const int mtiles  = (M + 15) >> 4;
  const int ngroups = N >> 6;
  const int waves   = mtiles * ngroups;
  const int blocks  = (waves + 7) / 8;      // 8 waves per 256-thread block
  const int ksteps  = K >> 5;
  if (ksteps == 16) {
    wmma_gemm_kernel<16><<<blocks, 256, 0, stream>>>(Ap, Bp, bias, C, M, N, ksteps);
  } else if (ksteps == 8) {
    wmma_gemm_kernel<8><<<blocks, 256, 0, stream>>>(Ap, Bp, bias, C, M, N, ksteps);
  } else {
    wmma_gemm_kernel<0><<<blocks, 256, 0, stream>>>(Ap, Bp, bias, C, M, N, ksteps);
  }
}

extern "C" void kernel_launch(void* const* d_in, const int* in_sizes, int n_in,
                              void* d_out, int out_size, void* d_ws, size_t ws_size,
                              hipStream_t stream) {
  const float* x_q   = (const float*)d_in[0];
  const float* w_mlp = (const float*)d_in[1];
  const float* b_mlp = (const float*)d_in[2];
  const float* w1    = (const float*)d_in[3];
  const float* b1    = (const float*)d_in[4];
  const float* w2    = (const float*)d_in[5];
  const float* b2    = (const float*)d_in[6];
  const int*   eidx  = (const int*)d_in[7];
  const int*   cs_u  = (const int*)d_in[8];
  const int*   cs_v  = (const int*)d_in[9];

  const int D_ENC = in_sizes[2];                 // 256
  const int D_IN  = in_sizes[1] / D_ENC;         // 512
  const int NQ    = in_sizes[0] / D_IN;          // 30000
  const int E     = in_sizes[7] / 2;             // 480000
  const int NPAIR = in_sizes[8];                 // 500000
  const int NT    = ((long)out_size - (long)NQ * D_ENC) / D_ENC;  // 60000

  const int* src = eidx;
  const int* dst = eidx + E;

  float* h_out = (float*)d_out;                        // [NQ x D_ENC]
  float* xt    = h_out + (size_t)NQ * D_ENC;           // [NT x D_ENC]

  // ---- workspace carve (256B aligned) ----
  char* ws = (char*)d_ws;
  size_t off = 0;
  auto carve = [&](size_t bytes) -> void* {
    void* p = ws + off;
    off += (bytes + 255) & ~(size_t)255;
    return p;
  };
  unsigned short* Ap  = (unsigned short*)carve((size_t)NQ * D_IN * 2);
  unsigned short* Wp  = (unsigned short*)carve((size_t)D_IN * D_ENC * 2);
  float*          B1  = (float*)carve((size_t)NQ * D_ENC * 4);
  float*          B2  = (float*)carve((size_t)NQ * D_ENC * 4);
  float*          dnv = (float*)carve((size_t)NQ * 4);
  float*          cnt = (float*)carve((size_t)NT * 4);
  (void)ws_size;

  const int mtiles = (NQ + 15) >> 4;

  // ---- degrees: deg = 1 (self loop) + in-degree; dinv = rsqrt ----
  fill_kernel<<<nblk(NQ, 256), 256, 0, stream>>>(dnv, 1.0f, NQ);
  deg_accum_kernel<<<nblk(E, 256), 256, 0, stream>>>(dst, dnv, E);
  rsqrt_kernel<<<nblk(NQ, 256), 256, 0, stream>>>(dnv, NQ);

  // ---- h0 = x_q @ w_mlp + b_mlp ----
  {
    long ta = (long)mtiles * (D_IN >> 5) * 512;
    long tw = (long)(D_ENC >> 4) * (D_IN >> 5) * 512;
    pack_a_kernel<<<nblk(ta, 256), 256, 0, stream>>>(x_q, Ap, NQ, D_IN);
    pack_w_kernel<<<nblk(tw, 256), 256, 0, stream>>>(w_mlp, Wp, D_IN, D_ENC);
    launch_gemm(Ap, Wp, b_mlp, B1, NQ, D_ENC, D_IN, stream);
  }

  long ta2 = (long)mtiles * (D_ENC >> 5) * 512;
  long tw2 = (long)(D_ENC >> 4) * (D_ENC >> 5) * 512;

  // ---- GCN layer 1: B1(h0) -> B2(h1) ----
  pack_a_kernel<<<nblk(ta2, 256), 256, 0, stream>>>(B1, Ap, NQ, D_ENC);
  pack_w_kernel<<<nblk(tw2, 256), 256, 0, stream>>>(w1, Wp, D_ENC, D_ENC);
  launch_gemm(Ap, Wp, nullptr, B2, NQ, D_ENC, D_ENC, stream);           // xw
  self_term_kernel<<<NQ, D_ENC, 0, stream>>>(B2, dnv, B1, D_ENC);       // agg=B1
  edge_agg_kernel<<<E, D_ENC / 4, 0, stream>>>(src, dst, B2, dnv, B1, D_ENC);
  bias_elu_kernel<<<NQ, D_ENC, 0, stream>>>(B1, b1, B2, D_ENC);         // h1=B2

  // ---- GCN layer 2: B2(h1) -> d_out h ----
  pack_a_kernel<<<nblk(ta2, 256), 256, 0, stream>>>(B2, Ap, NQ, D_ENC);
  pack_w_kernel<<<nblk(tw2, 256), 256, 0, stream>>>(w2, Wp, D_ENC, D_ENC);
  launch_gemm(Ap, Wp, nullptr, B1, NQ, D_ENC, D_ENC, stream);           // xw2
  self_term_kernel<<<NQ, D_ENC, 0, stream>>>(B1, dnv, B2, D_ENC);       // agg=B2
  edge_agg_kernel<<<E, D_ENC / 4, 0, stream>>>(src, dst, B1, dnv, B2, D_ENC);
  bias_elu_kernel<<<NQ, D_ENC, 0, stream>>>(B2, b2, h_out, D_ENC);      // final h

  // ---- pair scatter-mean into x_t ----
  fill_kernel<<<nblk(NT, 256), 256, 0, stream>>>(cnt, 0.0f, NT);
  count_pairs_kernel<<<nblk(NPAIR, 256), 256, 0, stream>>>(cs_v, cnt, NPAIR);
  fill_kernel<<<nblk((long)NT * D_ENC, 256), 256, 0, stream>>>(
      xt, 0.0f, (long)NT * D_ENC);
  pair_scatter_kernel<<<NPAIR, D_ENC / 4, 0, stream>>>(cs_u, cs_v, h_out, xt, D_ENC);
  finalize_xt_kernel<<<NT, D_ENC, 0, stream>>>(xt, cnt, D_ENC);
}
